// REGCNBase_58282706206737
// MI455X (gfx1250) — matrix operand order repaired and checked by reading, not compiled, not run on hardware
//
#include <hip/hip_runtime.h>
#include <hip/hip_bf16.h>

#define N_ENT   100000
#define N_REL2  1000
#define DIM     256
#define N_EDGE  200000
#define T_STEPS 3

typedef __attribute__((ext_vector_type(16))) __bf16 v16bf;
typedef __attribute__((ext_vector_type(8)))  __bf16 v8bf;
typedef __attribute__((ext_vector_type(8)))  float  v8f;

static __device__ __forceinline__ __bf16 f2bf(float x) { return (__bf16)x; }

// ---------------- h0 = l2norm(rows) ----------------
__global__ void l2norm_kernel(const float* __restrict__ in,
                              float* __restrict__ h, __bf16* __restrict__ hb) {
  __shared__ float red[8];
  const int n = blockIdx.x, d = threadIdx.x;
  const size_t i = (size_t)n * DIM + d;
  float v = in[i];
  float ss = v * v;
  for (int o = 16; o; o >>= 1) ss += __shfl_down(ss, o, 32);
  if ((d & 31) == 0) red[d >> 5] = ss;
  __syncthreads();
  if (d == 0) {
    float t = 0.f;
    for (int j = 0; j < 8; ++j) t += red[j];
    red[0] = fmaxf(sqrtf(t), 1e-12f);
  }
  __syncthreads();
  const float hv = v / red[0];
  h[i] = hv;
  hb[i] = f2bf(hv);
}

// ---------------- Wt[n][k] = bf16(W[k][n]) ----------------
__global__ void transpose_cvt_kernel(const float* __restrict__ W, __bf16* __restrict__ Wt) {
  const int n = blockIdx.x, k = threadIdx.x;
  Wt[(size_t)n * DIM + k] = f2bf(W[(size_t)k * DIM + n]);
}

// ---------------- fused GRU cell: out[n][d] ----------------
__global__ void gru_kernel(const float* __restrict__ srel, const float* __restrict__ crel,
                           const float* __restrict__ hprev,
                           const float* __restrict__ w_ih, const float* __restrict__ w_hh,
                           const float* __restrict__ b_ih, const float* __restrict__ b_hh,
                           float* __restrict__ out) {
  __shared__ float sx[2 * DIM];
  __shared__ float sh[DIM];
  const int n = blockIdx.x, d = threadIdx.x;
  sx[d]       = srel[(size_t)n * DIM + d];
  sx[DIM + d] = crel[(size_t)n * DIM + d];
  sh[d]       = hprev[(size_t)n * DIM + d];
  __syncthreads();
  float ir = b_ih[d], iz = b_ih[DIM + d], inn = b_ih[2 * DIM + d];
  const float* wr = w_ih + (size_t)d * (2 * DIM);
  const float* wz = w_ih + (size_t)(DIM + d) * (2 * DIM);
  const float* wn = w_ih + (size_t)(2 * DIM + d) * (2 * DIM);
  for (int k = 0; k < 2 * DIM; ++k) {
    const float xv = sx[k];
    ir += xv * wr[k]; iz += xv * wz[k]; inn += xv * wn[k];
  }
  float hr = b_hh[d], hz = b_hh[DIM + d], hn = b_hh[2 * DIM + d];
  const float* vr = w_hh + (size_t)d * DIM;
  const float* vz = w_hh + (size_t)(DIM + d) * DIM;
  const float* vn = w_hh + (size_t)(2 * DIM + d) * DIM;
  for (int k = 0; k < DIM; ++k) {
    const float hv = sh[k];
    hr += hv * vr[k]; hz += hv * vz[k]; hn += hv * vn[k];
  }
  const float r = 1.f / (1.f + __expf(-(ir + hr)));
  const float z = 1.f / (1.f + __expf(-(iz + hz)));
  const float nn = tanhf(inn + r * hn);
  out[(size_t)n * DIM + d] = (1.f - z) * nn + z * sh[d];
}

// ---------------- avg_rel: first-touch dedup + scatter ----------------
__global__ void avgrel_scatter_kernel(const int* __restrict__ edges, const float* __restrict__ h,
                                      unsigned* __restrict__ mask, float* __restrict__ sums,
                                      float* __restrict__ counts) {
  __shared__ int s_first, s_ent, s_rel;
  if (threadIdx.x == 0) {
    const int p = blockIdx.x;
    int ent, rel;
    if (p < N_EDGE) { ent = edges[(size_t)p * 3 + 0]; rel = edges[(size_t)p * 3 + 1]; }
    else { const int q = p - N_EDGE; ent = edges[(size_t)q * 3 + 2]; rel = edges[(size_t)q * 3 + 1]; }
    const unsigned code = (unsigned)ent * N_REL2 + (unsigned)rel;
    const unsigned bit = 1u << (code & 31u);
    const unsigned old = atomicOr(&mask[code >> 5], bit);
    const int first = (old & bit) == 0;
    if (first) atomicAdd(&counts[rel], 1.0f);
    s_first = first; s_ent = ent; s_rel = rel;
  }
  __syncthreads();
  if (s_first) {
    const float* hr = h + (size_t)s_ent * DIM;
    float* sr = sums + (size_t)s_rel * DIM;
    for (int i = threadIdx.x; i < DIM; i += 64) atomicAdd(&sr[i], hr[i]);
  }
}

__global__ void avgrel_finalize_kernel(const float* __restrict__ sums,
                                       const float* __restrict__ counts,
                                       float* __restrict__ cur_rel) {
  const int n = blockIdx.x, d = threadIdx.x;
  const float c = counts[n];
  const size_t i = (size_t)n * DIM + d;
  cur_rel[i] = (c > 0.f) ? (sums[i] / fmaxf(c, 1.f)) : 0.f;
}

// ---------------- degree histogram ----------------
__global__ void deg_kernel(const int* __restrict__ edges, float* __restrict__ deg) {
  const int e = blockIdx.x * blockDim.x + threadIdx.x;
  if (e < N_EDGE) atomicAdd(&deg[edges[(size_t)e * 3 + 2]], 1.0f);
}

// ---------------- RGCN pre-aggregation: pre[dst] += h[src] + rel[r] ----------------
__global__ void rgcn_scatter_kernel(const int* __restrict__ edges, const float* __restrict__ h,
                                    const float* __restrict__ rel_emb, float* __restrict__ pre) {
  const int e = blockIdx.x;
  const int src = edges[(size_t)e * 3 + 0];
  const int rel = edges[(size_t)e * 3 + 1];
  const int dst = edges[(size_t)e * 3 + 2];
  const float* hs = h + (size_t)src * DIM;
  const float* rr = rel_emb + (size_t)rel * DIM;
  float* pd = pre + (size_t)dst * DIM;
  for (int i = threadIdx.x; i < DIM; i += 64) atomicAdd(&pd[i], hs[i] + rr[i]);
}

// ---------------- scale by 1/deg + bf16 convert ----------------
__global__ void scale_cvt_kernel(const float* __restrict__ pre, const float* __restrict__ deg,
                                 __bf16* __restrict__ out) {
  const int row = blockIdx.x;
  const size_t i = (size_t)row * DIM + threadIdx.x;
  out[i] = f2bf(pre[i] * (1.0f / fmaxf(deg[row], 1.0f)));
}

__global__ void cvt_kernel(const float* __restrict__ x, __bf16* __restrict__ out) {
  const size_t i = (size_t)blockIdx.x * DIM + threadIdx.x;
  out[i] = f2bf(x[i]);
}

// ---------------- WMMA GEMM: C = A1@B1t^T (+ A2@B2t^T), bf16 in / f32 out ----------------
// B*t are column-major (pre-transposed) 256x256.
// Grid (M/16), block 128 (4 waves). Each wave: one 16-row tile x 4 n-tiles (16x64 of C),
// A fragments loaded once per K step and reused across the 4 accumulators.
template <bool DUAL>
__global__ void __launch_bounds__(128) wmma_gemm_kernel(
    const __bf16* __restrict__ A1, const __bf16* __restrict__ B1t,
    const __bf16* __restrict__ A2, const __bf16* __restrict__ B2t,
    float* __restrict__ C) {
  const int wave = threadIdx.x >> 5;     // 0..3
  const int lane = threadIdx.x & 31;
  const int half = lane & 15;
  const int sel  = lane >> 4;            // 0: lanes 0-15, 1: lanes 16-31
  const int mtile = blockIdx.x;
  const int ntile0 = wave * 4;           // 4 n-tiles per wave -> covers N=256 per block
  const size_t arow = (size_t)(mtile * 16 + half) * DIM;   // A row per lane
  const size_t brow = (size_t)half * DIM;                  // within-tile W^T row
  v8f acc[4] = {};
  for (int k = 0; k < DIM; k += 32) {
    const int aoff = k + 16 * sel;       // A elems j -> K = 16*sel + j
    const int boff = k + 16 * sel;       // B elems j -> K = 16*sel + j
    v8bf a1lo = *(const v8bf*)(A1 + arow + aoff);
    v8bf a1hi = *(const v8bf*)(A1 + arow + aoff + 8);
    const v16bf a1 = __builtin_shufflevector(a1lo, a1hi,
        0, 1, 2, 3, 4, 5, 6, 7, 8, 9, 10, 11, 12, 13, 14, 15);
    v16bf a2;
    if (DUAL) {
      v8bf a2lo = *(const v8bf*)(A2 + arow + aoff);
      v8bf a2hi = *(const v8bf*)(A2 + arow + aoff + 8);
      a2 = __builtin_shufflevector(a2lo, a2hi,
          0, 1, 2, 3, 4, 5, 6, 7, 8, 9, 10, 11, 12, 13, 14, 15);
    }
#pragma unroll
    for (int j = 0; j < 4; ++j) {
      const size_t bbase = (size_t)(ntile0 + j) * 16 * DIM + brow + boff;
      const v16bf b1 = *(const v16bf*)(B1t + bbase);
      acc[j] = __builtin_amdgcn_wmma_f32_16x16x32_bf16(
          false, a1, false, b1, (short)0, acc[j], false, false);
      if (DUAL) {
        const v16bf b2 = *(const v16bf*)(B2t + bbase);
        acc[j] = __builtin_amdgcn_wmma_f32_16x16x32_bf16(
            false, a2, false, b2, (short)0, acc[j], false, false);
      }
    }
  }
  const int rbase = mtile * 16 + 8 * sel;  // C VGPR v -> row v + 8*sel
#pragma unroll
  for (int j = 0; j < 4; ++j) {
    const int col = (ntile0 + j) * 16 + half;
#pragma unroll
    for (int v = 0; v < 8; ++v) C[(size_t)(rbase + v) * DIM + col] = acc[j][v];
  }
}

// ---------------- gate + blend ----------------
__global__ void combine_kernel(const float* __restrict__ G, const float* __restrict__ gb,
                               const float* __restrict__ cur2, float* __restrict__ h,
                               __bf16* __restrict__ hb) {
  const size_t i = (size_t)blockIdx.x * DIM + threadIdx.x;
  const float g = 1.f / (1.f + __expf(-(G[i] + gb[threadIdx.x])));
  const float hv = g * cur2[i] + (1.f - g) * h[i];
  h[i] = hv;
  hb[i] = f2bf(hv);
}

extern "C" void kernel_launch(void* const* d_in, const int* in_sizes, int n_in,
                              void* d_out, int out_size, void* d_ws, size_t ws_size,
                              hipStream_t stream) {
  (void)in_sizes; (void)n_in; (void)out_size; (void)ws_size;
  const int*   edges = (const int*)d_in[0];
  const float* ent   = (const float*)d_in[1];
  const float* srel  = (const float*)d_in[2];
  const float* gw    = (const float*)d_in[3];
  const float* gb    = (const float*)d_in[4];
  const float* w_ih  = (const float*)d_in[5];
  const float* w_hh  = (const float*)d_in[6];
  const float* b_ih  = (const float*)d_in[7];
  const float* b_hh  = (const float*)d_in[8];
  const float* w_nei = (const float*)d_in[9];
  const float* w_lp  = (const float*)d_in[10];

  float* H       = (float*)d_out;                        // 100000 x 256
  float* out_rel = H + (size_t)N_ENT * DIM;              // 1000 x 256

  const size_t NHf4 = (size_t)N_ENT * DIM * sizeof(float);
  const size_t NHf2 = (size_t)N_ENT * DIM * sizeof(__bf16);
  const size_t RELf4 = (size_t)N_REL2 * DIM * sizeof(float);
  const size_t Wf2 = (size_t)DIM * DIM * sizeof(__bf16);

  char* ws = (char*)d_ws;
  size_t off = 0;
  auto take = [&](size_t bytes) -> char* {
    char* p = ws + off;
    off += (bytes + 255) & ~(size_t)255;
    return p;
  };
  float*    PRE     = (float*)take(NHf4);
  float*    CUR     = (float*)take(NHf4);
  __bf16*   PREb    = (__bf16*)take(NHf2);
  __bf16*   CURb    = (__bf16*)take(NHf2);
  __bf16*   Hb      = (__bf16*)take(NHf2);
  unsigned* mask    = (unsigned*)take(((size_t)N_ENT * N_REL2 / 32) * 4);
  float*    deg     = (float*)take((size_t)N_ENT * 4);
  float*    relsums = (float*)take(RELf4);
  float*    relcnt  = (float*)take((size_t)N_REL2 * 4);
  float*    cur_rel = (float*)take(RELf4);
  float*    ERa     = (float*)take(RELf4);
  float*    ERb     = (float*)take(RELf4);
  __bf16*   Wn0t    = (__bf16*)take(Wf2);
  __bf16*   Wn1t    = (__bf16*)take(Wf2);
  __bf16*   Wl0t    = (__bf16*)take(Wf2);
  __bf16*   Wl1t    = (__bf16*)take(Wf2);
  __bf16*   Wgt     = (__bf16*)take(Wf2);

  // ---- prep ----
  l2norm_kernel<<<N_ENT, DIM, 0, stream>>>(ent, H, Hb);
  transpose_cvt_kernel<<<DIM, DIM, 0, stream>>>(w_nei,              Wn0t);
  transpose_cvt_kernel<<<DIM, DIM, 0, stream>>>(w_nei + DIM * DIM,  Wn1t);
  transpose_cvt_kernel<<<DIM, DIM, 0, stream>>>(w_lp,               Wl0t);
  transpose_cvt_kernel<<<DIM, DIM, 0, stream>>>(w_lp + DIM * DIM,   Wl1t);
  transpose_cvt_kernel<<<DIM, DIM, 0, stream>>>(gw,                 Wgt);

  // evolved_rel = gru([srel, 0], srel)
  hipMemsetAsync(cur_rel, 0, RELf4, stream);
  gru_kernel<<<N_REL2, DIM, 0, stream>>>(srel, cur_rel, srel, w_ih, w_hh, b_ih, b_hh, ERa);

  float* ERcur = ERa;
  float* ERnext = ERb;
  const dim3 ggrid(N_ENT / 16);

  for (int t = 0; t < T_STEPS; ++t) {
    const int* et = edges + (size_t)t * N_EDGE * 3;

    // ---- cur_rel = avg_rel_embed(edge, h) ----
    hipMemsetAsync(mask, 0, ((size_t)N_ENT * N_REL2 / 32) * 4, stream);
    hipMemsetAsync(relsums, 0, RELf4, stream);
    hipMemsetAsync(relcnt, 0, (size_t)N_REL2 * 4, stream);
    avgrel_scatter_kernel<<<2 * N_EDGE, 64, 0, stream>>>(et, H, mask, relsums, relcnt);
    avgrel_finalize_kernel<<<N_REL2, DIM, 0, stream>>>(relsums, relcnt, cur_rel);

    // ---- evolved_rel = gru([srel, cur_rel], evolved_rel) ----
    float* ERdst = (t == T_STEPS - 1) ? out_rel : ERnext;
    gru_kernel<<<N_REL2, DIM, 0, stream>>>(srel, cur_rel, ERcur, w_ih, w_hh, b_ih, b_hh, ERdst);
    float* ER = ERdst;

    // ---- degrees (shared by both layers) ----
    hipMemsetAsync(deg, 0, (size_t)N_ENT * 4, stream);
    deg_kernel<<<(N_EDGE + 255) / 256, 256, 0, stream>>>(et, deg);

    // ---- layer 0: CUR = (scatter(H)+rel)/deg @ Wn0 + H @ Wl0 ----
    hipMemsetAsync(PRE, 0, NHf4, stream);
    rgcn_scatter_kernel<<<N_EDGE, 64, 0, stream>>>(et, H, ER, PRE);
    scale_cvt_kernel<<<N_ENT, DIM, 0, stream>>>(PRE, deg, PREb);
    wmma_gemm_kernel<true><<<ggrid, 128, 0, stream>>>(PREb, Wn0t, Hb, Wl0t, CUR);
    cvt_kernel<<<N_ENT, DIM, 0, stream>>>(CUR, CURb);

    // ---- layer 1: PRE(f32 out) = (scatter(CUR)+rel)/deg @ Wn1 + CUR @ Wl1 ----
    hipMemsetAsync(PRE, 0, NHf4, stream);
    rgcn_scatter_kernel<<<N_EDGE, 64, 0, stream>>>(et, CUR, ER, PRE);
    scale_cvt_kernel<<<N_ENT, DIM, 0, stream>>>(PRE, deg, PREb);
    wmma_gemm_kernel<true><<<ggrid, 128, 0, stream>>>(PREb, Wn1t, CURb, Wl1t, PRE);

    // ---- gate: CUR = H @ gate_weight ----
    wmma_gemm_kernel<false><<<ggrid, 128, 0, stream>>>(Hb, Wgt, nullptr, nullptr, CUR);

    // ---- h = sigmoid(CUR + gb) * PRE + (1 - sigmoid) * h ----
    combine_kernel<<<N_ENT, DIM, 0, stream>>>(CUR, gb, PRE, H, Hb);

    if (t < T_STEPS - 1) { float* tmp = ERcur; ERcur = ERnext; ERnext = tmp; }
  }
}
